// MixtralGmmTop2MLP_24885040513468
// MI455X (gfx1250) — matrix operand order repaired and checked by reading, not compiled, not run on hardware
//
#include <hip/hip_runtime.h>
#include <hip/hip_bf16.h>
#include <stdint.h>

// ---------------- problem constants ----------------
#define TOKENS 32768
#define TOPK   2
#define ROWS   (TOKENS*TOPK)      // 65536 (token, slot) rows
#define HID    2048
#define FFN    4096
#define NE     8
#define BM     128                // rows per tile
#define BN     64                 // cols per tile
#define KC     32                 // K chunk (WMMA bf16 K)
#define MAX_TILES (ROWS/BM + NE)  // 520 upper bound on row tiles
#define ROWS_PAD  (ROWS + NE*BM)  // padded sorted-row space (66560)

typedef unsigned short u16;
typedef __bf16 v16bf __attribute__((ext_vector_type(16)));
typedef __bf16 v8bf  __attribute__((ext_vector_type(8)));
typedef float  v8f   __attribute__((ext_vector_type(8)));
typedef int    v4i   __attribute__((ext_vector_type(4)));

// Async global->LDS path (CDNA5). Guarded: fall back to plain loads if the
// builtins are absent on this toolchain.
#if defined(__HIP_DEVICE_COMPILE__) && \
    __has_builtin(__builtin_amdgcn_global_load_async_to_lds_b128) && \
    __has_builtin(__builtin_amdgcn_s_wait_asynccnt)
#define USE_ASYNC 1
#else
#define USE_ASYNC 0
#endif

__device__ __forceinline__ u16 f2bf(float f) {           // RNE float->bf16 bits
  unsigned u = __float_as_uint(f);
  u += 0x7FFFu + ((u >> 16) & 1u);
  return (u16)(u >> 16);
}

__device__ __forceinline__ void cp16(const u16* g, u16* l) {
#if USE_ASYNC
  // (v4i addrspace(1)*, v4i addrspace(3)*, imm offset, imm cpol)
  __builtin_amdgcn_global_load_async_to_lds_b128(
      (__attribute__((address_space(1))) v4i*)(g),
      (__attribute__((address_space(3))) v4i*)(l), 0, 0);
#else
  *(uint4*)l = *(const uint4*)g;
#endif
}

__device__ __forceinline__ v16bf mk16(v8bf lo, v8bf hi) {
  return __builtin_shufflevector(lo, hi, 0,1,2,3,4,5,6,7,8,9,10,11,12,13,14,15);
}
// A-frag: per lane K = {kb..kb+7, kb+16..kb+23}, p points at kb within a 32-K row
__device__ __forceinline__ v16bf ld_a_frag(const u16* p) {
  return mk16(*(const v8bf*)p, *(const v8bf*)(p + 16));
}
// B-frag: per lane K = {kb..kb+7, kb+16..kb+23}, p points at kb within a 32-K row
__device__ __forceinline__ v16bf ld_b_frag(const u16* p) {
  return mk16(*(const v8bf*)p, *(const v8bf*)(p + 16));
}
#define WMMA_BF16(a,b,c) \
  __builtin_amdgcn_wmma_f32_16x16x32_bf16(false,(a),false,(b),(short)0,(c),false,false)

// ---------------- routing ----------------
// meta layout: [0..7]=counts [8..15]=poff [16..23]=cursor [24]=n_tiles
__global__ void k_zero(int* meta) {
  if (threadIdx.x < 32) meta[threadIdx.x] = 0;
}

__global__ void k_hist(const int* __restrict__ topk, int* __restrict__ meta) {
  int i = blockIdx.x * blockDim.x + threadIdx.x;
  int stride = gridDim.x * blockDim.x;
  for (; i < ROWS; i += stride) atomicAdd(&meta[topk[i]], 1);
}

__global__ void k_prep(int* meta, int* row_map, int* tile_e, int* tile_r) {
  if (threadIdx.x == 0 && blockIdx.x == 0) {
    int off = 0, tiles = 0;
    for (int e = 0; e < NE; ++e) {
      int cnt = meta[e];
      meta[8 + e] = off;
      int nt = (cnt + BM - 1) / BM;
      for (int t = 0; t < nt; ++t) { tile_e[tiles] = e; tile_r[tiles] = off + t*BM; ++tiles; }
      for (int p = cnt; p < nt*BM; ++p) row_map[off + p] = -1;   // pad rows
      off += nt * BM;
    }
    meta[24] = tiles;
  }
}

__global__ void k_scatter(const int* __restrict__ topk, int* __restrict__ meta,
                          int* __restrict__ row_map) {
  int i = blockIdx.x * blockDim.x + threadIdx.x;
  int stride = gridDim.x * blockDim.x;
  for (; i < ROWS; i += stride) {
    int e = topk[i];
    int p = atomicAdd(&meta[16 + e], 1);
    row_map[meta[8 + e] + p] = i;       // i = m*TOPK + k == output row index
  }
}

// ---------------- precision conversion ----------------
__global__ void k_cvt(const float* __restrict__ s, u16* __restrict__ d, long n4) {
  long i = (long)blockIdx.x * blockDim.x + threadIdx.x;
  long stride = (long)gridDim.x * blockDim.x;
  for (; i < n4; i += stride) {
    float4 v = ((const float4*)s)[i];
    ushort4 o;
    o.x = f2bf(v.x); o.y = f2bf(v.y); o.z = f2bf(v.z); o.w = f2bf(v.w);
    ((ushort4*)d)[i] = o;
  }
}

// transpose [R][C] fp32 -> [C][R] bf16 (per batch slice), block (32,8)
__global__ void k_tconv(const float* __restrict__ src, u16* __restrict__ dst, int R, int C) {
  __shared__ float t[32][33];
  const float* s = src + (size_t)blockIdx.z * R * C;
  u16* d = dst + (size_t)blockIdx.z * R * C;
  int c0 = blockIdx.x * 32, r0 = blockIdx.y * 32;
  for (int j = 0; j < 32; j += 8)
    t[threadIdx.y + j][threadIdx.x] = s[(size_t)(r0 + threadIdx.y + j) * C + c0 + threadIdx.x];
  __syncthreads();
  for (int j = 0; j < 32; j += 8)
    d[(size_t)(c0 + threadIdx.y + j) * R + r0 + threadIdx.x] = f2bf(t[threadIdx.x][threadIdx.y + j]);
}

// ---------------- GEMM1: h = silu(x@w1) * (x@w3), bf16 in/out, f32 acc ----------------
// 128x64 block tile, 8 waves in a 4(M-pairs) x 2(N-pairs) grid, 2x2 16x16 tiles/wave.
__global__ __launch_bounds__(256) void k_gemm1(
    const u16* __restrict__ xb, const u16* __restrict__ w1t, const u16* __restrict__ w3t,
    u16* __restrict__ hbuf, const int* __restrict__ row_map,
    const int* __restrict__ tile_e, const int* __restrict__ tile_r,
    const int* __restrict__ meta)
{
  if ((int)blockIdx.x >= meta[24]) return;
  const int e = tile_e[blockIdx.x];
  const int r0 = tile_r[blockIdx.x];
  const int nbase = blockIdx.y * BN;

  __shared__ __align__(16) u16 As[2][BM * KC];   // 8 KB per buffer
  __shared__ __align__(16) u16 B1s[2][BN * KC];  // 4 KB per buffer
  __shared__ __align__(16) u16 B3s[2][BN * KC];

  const int tid = threadIdx.x;
  const int lr = tid >> 2, ls = tid & 3;         // segment (row/col, 16B chunk)
  const int st0 = lr * KC + ls * 8;              // A rows lr and lr+64
  const int st1 = st0 + 64 * KC;

  const int srow0 = row_map[r0 + lr];
  const int srow1 = row_map[r0 + lr + 64];
  const int tok0 = srow0 >= 0 ? (srow0 >> 1) : 0;   // pad rows read token 0
  const int tok1 = srow1 >= 0 ? (srow1 >> 1) : 0;
  const u16* gA0 = xb + (size_t)tok0 * HID + ls * 8;
  const u16* gA1 = xb + (size_t)tok1 * HID + ls * 8;
  const u16* gB1 = w1t + ((size_t)e * FFN + nbase + lr) * HID + ls * 8;  // n-major
  const u16* gB3 = w3t + ((size_t)e * FFN + nbase + lr) * HID + ls * 8;

  const int lane = tid & 31, l16 = lane & 15, kh = lane >> 4;
  const int wv = tid >> 5, miq = wv & 3, nq = wv >> 2;
  const int aoff0 = ((miq * 2 + 0) * 16 + l16) * KC + kh * 8;
  const int aoff1 = ((miq * 2 + 1) * 16 + l16) * KC + kh * 8;
  const int boff0 = ((nq * 2 + 0) * 16 + l16) * KC + kh * 8;
  const int boff1 = ((nq * 2 + 1) * 16 + l16) * KC + kh * 8;

  v8f c1_00 = {}, c1_01 = {}, c1_10 = {}, c1_11 = {};
  v8f c3_00 = {}, c3_01 = {}, c3_10 = {}, c3_11 = {};
  const int NK = HID / KC;  // 64

  cp16(gA0, &As[0][st0]);
  cp16(gA1, &As[0][st1]);
  cp16(gB1, &B1s[0][lr * KC + ls * 8]);
  cp16(gB3, &B3s[0][lr * KC + ls * 8]);

  for (int ks = 0; ks < NK; ++ks) {
    const int cur = ks & 1;
    if (ks + 1 < NK) {
      const int nxt = cur ^ 1;
      const size_t go = (size_t)(ks + 1) * KC;
      cp16(gA0 + go, &As[nxt][st0]);
      cp16(gA1 + go, &As[nxt][st1]);
      cp16(gB1 + go, &B1s[nxt][lr * KC + ls * 8]);
      cp16(gB3 + go, &B3s[nxt][lr * KC + ls * 8]);
#if USE_ASYNC
      __builtin_amdgcn_s_wait_asynccnt(4);   // current batch drained, next in flight
#endif
    } else {
#if USE_ASYNC
      __builtin_amdgcn_s_wait_asynccnt(0);
#endif
    }
    __syncthreads();
    v16bf a0  = ld_a_frag(&As[cur][aoff0]);
    v16bf a1  = ld_a_frag(&As[cur][aoff1]);
    v16bf b10 = ld_b_frag(&B1s[cur][boff0]);
    v16bf b11 = ld_b_frag(&B1s[cur][boff1]);
    v16bf b30 = ld_b_frag(&B3s[cur][boff0]);
    v16bf b31 = ld_b_frag(&B3s[cur][boff1]);
    c1_00 = WMMA_BF16(a0, b10, c1_00);
    c1_01 = WMMA_BF16(a0, b11, c1_01);
    c1_10 = WMMA_BF16(a1, b10, c1_10);
    c1_11 = WMMA_BF16(a1, b11, c1_11);
    c3_00 = WMMA_BF16(a0, b30, c3_00);
    c3_01 = WMMA_BF16(a0, b31, c3_01);
    c3_10 = WMMA_BF16(a1, b30, c3_10);
    c3_11 = WMMA_BF16(a1, b31, c3_11);
    __syncthreads();
  }

#pragma unroll
  for (int im = 0; im < 2; ++im) {
#pragma unroll
    for (int jn = 0; jn < 2; ++jn) {
      v8f A1 = im ? (jn ? c1_11 : c1_10) : (jn ? c1_01 : c1_00);
      v8f A3 = im ? (jn ? c3_11 : c3_10) : (jn ? c3_01 : c3_00);
      const int col = nbase + (nq * 2 + jn) * 16 + l16;
      const int rowb = r0 + (miq * 2 + im) * 16 + kh * 8;
#pragma unroll
      for (int v = 0; v < 8; ++v) {
        float g = A1[v];
        float hval = (g / (1.0f + __expf(-g))) * A3[v];  // silu(gate) * up
        hbuf[(size_t)(rowb + v) * FFN + col] = f2bf(hval);
      }
    }
  }
}

// ---------------- GEMM2: out = h @ w2, scatter back by row_map ----------------
__global__ __launch_bounds__(256) void k_gemm2(
    const u16* __restrict__ hbuf, const u16* __restrict__ w2t,
    float* __restrict__ out, const int* __restrict__ row_map,
    const int* __restrict__ tile_e, const int* __restrict__ tile_r,
    const int* __restrict__ meta)
{
  if ((int)blockIdx.x >= meta[24]) return;
  const int e = tile_e[blockIdx.x];
  const int r0 = tile_r[blockIdx.x];
  const int nbase = blockIdx.y * BN;

  __shared__ __align__(16) u16 As[2][BM * KC];
  __shared__ __align__(16) u16 Bs[2][BN * KC];

  const int tid = threadIdx.x;
  const int lr = tid >> 2, ls = tid & 3;
  const int st0 = lr * KC + ls * 8;
  const int st1 = st0 + 64 * KC;
  const u16* gA0 = hbuf + (size_t)(r0 + lr) * FFN + ls * 8;            // sorted rows
  const u16* gA1 = hbuf + (size_t)(r0 + lr + 64) * FFN + ls * 8;
  const u16* gB  = w2t + ((size_t)e * HID + nbase + lr) * FFN + ls * 8; // n-major

  const int lane = tid & 31, l16 = lane & 15, kh = lane >> 4;
  const int wv = tid >> 5, miq = wv & 3, nq = wv >> 2;
  const int aoff0 = ((miq * 2 + 0) * 16 + l16) * KC + kh * 8;
  const int aoff1 = ((miq * 2 + 1) * 16 + l16) * KC + kh * 8;
  const int boff0 = ((nq * 2 + 0) * 16 + l16) * KC + kh * 8;
  const int boff1 = ((nq * 2 + 1) * 16 + l16) * KC + kh * 8;

  v8f c00 = {}, c01 = {}, c10 = {}, c11 = {};
  const int NK = FFN / KC;  // 128

  cp16(gA0, &As[0][st0]);
  cp16(gA1, &As[0][st1]);
  cp16(gB,  &Bs[0][st0]);

  for (int ks = 0; ks < NK; ++ks) {
    const int cur = ks & 1;
    if (ks + 1 < NK) {
      const int nxt = cur ^ 1;
      const size_t go = (size_t)(ks + 1) * KC;
      cp16(gA0 + go, &As[nxt][st0]);
      cp16(gA1 + go, &As[nxt][st1]);
      cp16(gB  + go, &Bs[nxt][st0]);
#if USE_ASYNC
      __builtin_amdgcn_s_wait_asynccnt(3);
#endif
    } else {
#if USE_ASYNC
      __builtin_amdgcn_s_wait_asynccnt(0);
#endif
    }
    __syncthreads();
    v16bf a0 = ld_a_frag(&As[cur][aoff0]);
    v16bf a1 = ld_a_frag(&As[cur][aoff1]);
    v16bf b0 = ld_b_frag(&Bs[cur][boff0]);
    v16bf b1 = ld_b_frag(&Bs[cur][boff1]);
    c00 = WMMA_BF16(a0, b0, c00);
    c01 = WMMA_BF16(a0, b1, c01);
    c10 = WMMA_BF16(a1, b0, c10);
    c11 = WMMA_BF16(a1, b1, c11);
    __syncthreads();
  }

#pragma unroll
  for (int im = 0; im < 2; ++im) {
#pragma unroll
    for (int jn = 0; jn < 2; ++jn) {
      v8f A = im ? (jn ? c11 : c10) : (jn ? c01 : c00);
      const int col = nbase + (nq * 2 + jn) * 16 + l16;
      const int rowb = r0 + (miq * 2 + im) * 16 + kh * 8;
#pragma unroll
      for (int v = 0; v < 8; ++v) {
        int srow = row_map[rowb + v];
        if (srow >= 0) out[(size_t)srow * HID + col] = A[v];  // unique writer per row
      }
    }
  }
}

// ---------------- launcher ----------------
extern "C" void kernel_launch(void* const* d_in, const int* in_sizes, int n_in,
                              void* d_out, int out_size, void* d_ws, size_t ws_size,
                              hipStream_t stream)
{
  const float* x   = (const float*)d_in[0];
  const int*   topk= (const int*)d_in[1];
  const float* w1  = (const float*)d_in[2];
  const float* w2  = (const float*)d_in[3];
  const float* w3  = (const float*)d_in[4];
  float* out = (float*)d_out;

  // Workspace carve-out (~1.08 GB total)
  char* ws = (char*)d_ws;
  const size_t SZ_XB  = (size_t)TOKENS * HID * 2;   // 128 MB
  const size_t SZ_W13 = (size_t)NE * FFN * HID * 2; // 128 MB each
  const size_t SZ_H   = (size_t)ROWS_PAD * FFN * 2; // ~520 MB
  u16* xb   = (u16*)(ws);
  u16* w1t  = (u16*)(ws + SZ_XB);
  u16* w3t  = (u16*)(ws + SZ_XB + SZ_W13);
  u16* w2t  = (u16*)(ws + SZ_XB + 2 * SZ_W13);
  u16* hbuf = (u16*)(ws + SZ_XB + 3 * SZ_W13);
  int* row_map = (int*)(ws + SZ_XB + 3 * SZ_W13 + SZ_H);
  int* meta   = row_map + ROWS_PAD;
  int* tile_e = meta + 32;
  int* tile_r = tile_e + MAX_TILES;

  // routing
  k_zero<<<1, 32, 0, stream>>>(meta);
  k_hist<<<256, 256, 0, stream>>>(topk, meta);
  k_prep<<<1, 1, 0, stream>>>(meta, row_map, tile_e, tile_r);
  k_scatter<<<256, 256, 0, stream>>>(topk, meta, row_map);

  // fp32 -> bf16 (x row-major; weights transposed to n-major)
  k_cvt<<<8192, 256, 0, stream>>>(x, xb, (long)TOKENS * HID / 4);
  dim3 tb(32, 8);
  k_tconv<<<dim3(FFN/32, HID/32, NE), tb, 0, stream>>>(w1, w1t, HID, FFN);
  k_tconv<<<dim3(FFN/32, HID/32, NE), tb, 0, stream>>>(w3, w3t, HID, FFN);
  k_tconv<<<dim3(HID/32, FFN/32, NE), tb, 0, stream>>>(w2, w2t, FFN, HID);

  // grouped GEMMs on WMMA
  k_gemm1<<<dim3(MAX_TILES, FFN/BN), 256, 0, stream>>>(xb, w1t, w3t, hbuf,
                                                       row_map, tile_e, tile_r, meta);
  k_gemm2<<<dim3(MAX_TILES, HID/BN), 256, 0, stream>>>(hbuf, w2t, out,
                                                       row_map, tile_e, tile_r, meta);
}
